// HanDecoder_25950192403231
// MI455X (gfx1250) — matrix operand order repaired and checked by reading, not compiled, not run on hardware
//
#include <hip/hip_runtime.h>

// ---------------- problem constants ----------------
#define B_      16
#define N_      2048
#define NOTE_   512
#define OUT_    11
#define D_      10          // OUT-1
#define G_      512         // 4*128 gate width (both LSTMs)
#define KP_T    224         // tempo input 203 padded to 7*32
#define KT_T    7
#define KP_F    736         // final input 715 padded to 23*32
#define KT_F    23
#define KT_H    4           // hidden 128 = 4*32
#define NT_     32          // 512/16 N-tiles

typedef __attribute__((ext_vector_type(8)))  _Float16 v8h;
typedef __attribute__((ext_vector_type(16))) _Float16 v16h;
typedef __attribute__((ext_vector_type(8)))  float    v8f;

// global-address-space vector pointer -> forces global_load_b128 (not flat_load)
typedef const __attribute__((address_space(1))) v8h* g_v8h_p;

__device__ __forceinline__ float sigf(float x) { return 1.0f / (1.0f + __expf(-x)); }

__device__ __forceinline__ v16h cat16(v8h lo, v8h hi) {
  return __builtin_shufflevector(lo, hi, 0,1,2,3,4,5,6,7,8,9,10,11,12,13,14,15);
}

// A fragment (16x32 f16) from row-major LDS activations [16 x stride].
// lane L: row m=L&15; halves 0..7 -> K = kt*32 + 8*(L>>4) + 0..7 ; halves 8..15 -> +16.
__device__ __forceinline__ v16h load_a(const _Float16* act, int stride, int m, int hs, int kt) {
  const _Float16* p = act + m * stride + kt * 32 + hs * 8;
  v8h lo = *(const v8h*)p;
  v8h hi = *(const v8h*)(p + 16);
  return cat16(lo, hi);
}

// B fragment (32x16 f16) from packed weights in global memory:
// 512 halves/tile, each lane reads its 32 contiguous bytes (two b128 loads).
__device__ __forceinline__ v16h load_b(unsigned long long base, int tile, int lane) {
  const unsigned long long addr =
      base + ((unsigned long long)tile * 512ull + (unsigned long long)(lane * 16)) * 2ull;
  g_v8h_p p = (g_v8h_p)addr;
  v8h lo = p[0];
  v8h hi = p[1];
  return cat16(lo, hi);
}

__device__ __forceinline__ v8f wmma_f16(v16h a, v16h b, v8f c) {
  return __builtin_amdgcn_wmma_f32_16x16x32_f16(false, a, false, b, (short)0, c, false, false);
}

// gates[16 x 512] = act(16 x K) @ packIh + h16(16 x 128) @ packHh + bias
// Bias is folded into the WMMA C initializer (one N-column per lane).
template<int KT_IH>
__device__ __forceinline__ void gemm_gates(const _Float16* actIh, int strideIh,
                                           const _Float16* hmat,
                                           const _Float16* packIh, const _Float16* packHh,
                                           const float* biasL, float* gates,
                                           int lane, int w) {
  // defeat LICM across the 2048-step loop: keep weight tiles as per-step
  // global (L2-resident) reloads; "+s" keeps the base uniform in SGPRs so the
  // loads lower to global_load_b128 saddr form.
  unsigned long long ihB = (unsigned long long)packIh;
  unsigned long long hhB = (unsigned long long)packHh;
  asm volatile("" : "+s"(ihB), "+s"(hhB));
  const int nt0 = w * 2, nt1 = nt0 + 1;
  const int m = lane & 15, hs = lane >> 4;
  const int col0 = nt0 * 16 + m, col1 = nt1 * 16 + m;
  const float b0 = biasL[col0], b1 = biasL[col1];
  v8f acc0, acc1;
#pragma unroll
  for (int r = 0; r < 8; ++r) { acc0[r] = b0; acc1[r] = b1; }
#pragma unroll
  for (int kt = 0; kt < KT_IH; ++kt) {
    v16h a = load_a(actIh, strideIh, m, hs, kt);
    acc0 = wmma_f16(a, load_b(ihB, kt * NT_ + nt0, lane), acc0);
    acc1 = wmma_f16(a, load_b(ihB, kt * NT_ + nt1, lane), acc1);
  }
#pragma unroll
  for (int kt = 0; kt < KT_H; ++kt) {
    v16h a = load_a(hmat, 128, m, hs, kt);
    acc0 = wmma_f16(a, load_b(hhB, kt * NT_ + nt0, lane), acc0);
    acc1 = wmma_f16(a, load_b(hhB, kt * NT_ + nt1, lane), acc1);
  }
  // C/D layout: VGPR r, lanes0-15 -> M=r, lanes16-31 -> M=8+r; N = nt*16 + (lane&15)
#pragma unroll
  for (int r = 0; r < 8; ++r) {
    const int br = r + 8 * hs;
    gates[br * G_ + col0] = acc0[r];
    gates[br * G_ + col1] = acc1[r];
  }
}

// ---------------- prep kernels ----------------
// Pack W (512 x Kact, row-major) as WMMA B tiles of W^T (Kpad x 512), zero-padded K.
__global__ void pack_wmma_b(const float* __restrict__ W, _Float16* __restrict__ dst, int Kact) {
  const int t = blockIdx.x;          // tile = kt*NT_ + nt
  const int lane = threadIdx.x;      // 0..31
  const int nt = t % NT_;
  const int kt = t / NT_;
  const int n  = nt * 16 + (lane & 15);
  const int kb = kt * 32 + 8 * (lane >> 4);
  _Float16* o = dst + (size_t)t * 512 + lane * 16;
#pragma unroll
  for (int h = 0; h < 16; ++h) {
    const int k = kb + h + (h & 8);
    o[h] = (_Float16)((k < Kact) ? W[(size_t)n * Kact + k] : 0.f);
  }
}

__global__ void prep_misc(const float* __restrict__ bih_t, const float* __restrict__ bhh_t,
                          const float* __restrict__ bih_f, const float* __restrict__ bhh_f,
                          const float* __restrict__ Wa, const float* __restrict__ ba,
                          const float* __restrict__ ctx,
                          float* __restrict__ biasT, float* __restrict__ biasF,
                          float* __restrict__ attnV) {
  const int t = blockIdx.x * blockDim.x + threadIdx.x;
  if (t < 512) biasT[t] = bih_t[t] + bhh_t[t];
  else if (t < 1024) biasF[t - 512] = bih_f[t - 512] + bhh_f[t - 512];
  else if (t < 1024 + D_) {
    const int k = t - 1024;
    float s = 0.f;
    for (int j = 0; j < D_; ++j) s += ctx[j] * Wa[j * D_ + k];   // v = Wa^T ctx
    attnV[k] = s;
  } else if (t == 1024 + D_) {
    float s = 0.f;
    for (int j = 0; j < D_; ++j) s += ctx[j] * ba[j];            // ctx . ba
    attnV[D_] = s;
  }
}

// ---------------- persistent sequential decoder ----------------
// One workgroup (16 wave32 = 512 threads) on one WGP. Wave w owns batch row w.
// Recurrent LSTM state lives in registers (4 elems/lane); only data that must
// cross waves (activation rows, gate matrix, f16 h mirrors) lives in LDS.
__global__ void __launch_bounds__(512, 1)
han_decoder_seq(const float* __restrict__ note_emb,
                const float* __restrict__ beat_emb,
                const float* __restrict__ measure_emb,
                const int*   __restrict__ beat_numbers,
                const int*   __restrict__ measure_numbers,
                const float* __restrict__ W_fc, const float* __restrict__ b_fc,
                const float* __restrict__ W_tfc, const float* __restrict__ b_tfc,
                const _Float16* __restrict__ wtP, const _Float16* __restrict__ whtP,
                const _Float16* __restrict__ wfP, const _Float16* __restrict__ whfP,
                const float* __restrict__ biasT, const float* __restrict__ biasF,
                const float* __restrict__ attnV, float* __restrict__ padF,
                float* __restrict__ out) {
  extern __shared__ char smem_raw[];
  float* gates  = (float*)smem_raw;            // 16*512
  float* prevo  = gates  + B_ * G_;            // 16*11 (slot0 mirrored in regs)
  float* bcur   = prevo  + B_ * OUT_;          // 16*10 current-beat result_node
  float* WfcL   = bcur   + B_ * D_;            // 10*128
  float* WtfcL  = WfcL   + D_ * 128;           // 128
  float* biasTL = WtfcL  + 128;                // 512
  float* biasFL = biasTL + G_;                 // 512
  _Float16* h16  = (_Float16*)(biasFL + G_);   // 16*128 tempo-h (WMMA A)
  _Float16* fh16 = h16  + B_ * 128;            // 16*128 final-h (WMMA A)
  _Float16* actT = fh16 + B_ * 128;            // 16*224
  _Float16* actF = actT + B_ * KP_T;           // 16*736

  const int tid  = threadIdx.x;
  const int lane = tid & 31;
  const int w    = tid >> 5;
  const int b    = w;                          // wave <-> batch row

  // ---- one-time staging ----
  for (int idx = tid; idx < D_ * 128; idx += 512) WfcL[idx] = W_fc[idx];
  if (tid < 128) WtfcL[tid] = W_tfc[tid];
  if (tid < G_) { biasTL[tid] = biasT[tid]; biasFL[tid] = biasF[tid]; }
  for (int idx = tid; idx < B_ * 128; idx += 512) {
    h16[idx] = (_Float16)0.f; fh16[idx] = (_Float16)0.f;
  }
  if (tid < B_ * OUT_) prevo[tid] = 0.f;
  if (tid < B_ * D_)   bcur[tid]  = 0.f;

  float av[D_], bfcR[D_];
#pragma unroll
  for (int k = 0; k < D_; ++k) { av[k] = attnV[k]; bfcR[k] = b_fc[k]; }
  const float attnC = attnV[D_];
  const float btfc  = b_tfc[0];

  float th_r[4], tc_r[4], fc_r[4];
#pragma unroll
  for (int q = 0; q < 4; ++q) { th_r[q] = 0.f; tc_r[q] = 0.f; fc_r[q] = 0.f; }
  float prevo0 = 0.f;                           // prev_out[:,0] (uniform per wave)
  int runStart = 0;                             // start of current beat run
  const int bn0 = beat_numbers[b * N_];
  const int mn0 = measure_numbers[b * N_];
  __syncthreads();

  for (int i = 0; i < N_; ++i) {
    // ===== SA: flags + lane-parallel intra-beat attention + build actT row =====
    const int bn_i = beat_numbers[b * N_ + i];
    const int bn_p = beat_numbers[b * N_ + (i > 0 ? i - 1 : 0)];
    const int ch = (i == 0) ? 1 : (bn_i > bn_p ? 1 : 0);
    const int cb = bn_i - bn0;
    const int cm = measure_numbers[b * N_ + i] - mn0;

    float mx = -1e30f, den = 0.f, acc[D_];
#pragma unroll
    for (int k = 0; k < D_; ++k) acc[k] = 0.f;
    for (int j = runStart + lane; j < i; j += 32) {      // online softmax per lane
      const float* xr = out + (size_t)(b * N_ + j) * OUT_ + 1;
      float x[D_];
#pragma unroll
      for (int k = 0; k < D_; ++k) x[k] = xr[k];
      float sim = attnC;
#pragma unroll
      for (int k = 0; k < D_; ++k) sim += av[k] * x[k];
      if (sim > mx) {
        const float sc = __expf(mx - sim);
        den *= sc;
#pragma unroll
        for (int k = 0; k < D_; ++k) acc[k] *= sc;
        mx = sim;
      }
      const float e = __expf(sim - mx);
      den += e;
#pragma unroll
      for (int k = 0; k < D_; ++k) acc[k] += e * x[k];
    }
#pragma unroll
    for (int off = 16; off > 0; off >>= 1) {             // butterfly combine
      const float m2 = __shfl_xor(mx, off, 32);
      const float d2 = __shfl_xor(den, off, 32);
      float a2[D_];
#pragma unroll
      for (int k = 0; k < D_; ++k) a2[k] = __shfl_xor(acc[k], off, 32);
      const float mn = fmaxf(mx, m2);
      const float s1 = __expf(mx - mn), s2 = __expf(m2 - mn);
      den = den * s1 + d2 * s2;
#pragma unroll
      for (int k = 0; k < D_; ++k) acc[k] = acc[k] * s1 + a2[k] * s2;
      mx = mn;
    }
    if (ch) {
      const float inv = (i > runStart) ? (1.f / den) : 0.f;  // empty mask -> 0
      if (lane == 0) {
#pragma unroll
        for (int k = 0; k < D_; ++k) bcur[b * D_ + k] = acc[k] * inv;
      }
      if (i > 0) runStart = i;
    }
    __builtin_amdgcn_wave_barrier();   // keep lane0 LDS store before wave's loads

    // tempo input row (wave-local): [beat 128 | measure 64 | tempo 1 | beat_res 10]
    for (int k = lane; k < KP_T; k += 32) {
      float v;
      if      (k < 128)  v = beat_emb[(size_t)(b * 256 + cb) * 128 + k];
      else if (k < 192)  v = measure_emb[(size_t)(b * 64 + cm) * 64 + (k - 128)];
      else if (k == 192) v = prevo0;
      else if (k < 203)  v = bcur[b * D_ + (k - 193)];
      else               v = 0.f;
      actT[b * KP_T + k] = (_Float16)v;
    }
    __syncthreads();

    // ===== SB: tempo gates via WMMA =====
    gemm_gates<KT_T>(actT, KP_T, h16, wtP, whtP, biasTL, gates, lane, w);
    __syncthreads();

    // ===== SC: tempo LSTM (regs) + tempo head + build actF row =====
    float h2v[4];
#pragma unroll
    for (int q = 0; q < 4; ++q) {
      const int j = lane + 32 * q;
      const float ii = gates[b * G_ + j],       ff = gates[b * G_ + 128 + j];
      const float gg = gates[b * G_ + 256 + j], oo = gates[b * G_ + 384 + j];
      const float c2 = sigf(ff) * tc_r[q] + sigf(ii) * tanhf(gg);
      const float h2 = sigf(oo) * tanhf(c2);
      h2v[q] = h2;
      const float nh = ch ? h2 : th_r[q];
      const float nc = ch ? c2 : tc_r[q];
      th_r[q] = nh; tc_r[q] = nc;
      h16[b * 128 + j] = (_Float16)nh;
    }
    float part = 0.f;
#pragma unroll
    for (int q = 0; q < 4; ++q) part += h2v[q] * WtfcL[lane + 32 * q];
#pragma unroll
    for (int off = 16; off > 0; off >>= 1) part += __shfl_xor(part, off, 32);
    prevo0 = ch ? (part + btfc) : prevo0;

    // final input row: [note 512 | beat 128 | measure 64 | prev_out 11];
    // fold in the pad-mask row sum while the note row streams through.
    float nsum = 0.f;
    for (int k = lane; k < KP_F; k += 32) {
      float v;
      if      (k < 512)  { v = note_emb[(size_t)(b * N_ + i) * 512 + k]; nsum += v; }
      else if (k < 640)  v = beat_emb[(size_t)(b * 256 + cb) * 128 + (k - 512)];
      else if (k < 704)  v = measure_emb[(size_t)(b * 64 + cm) * 64 + (k - 640)];
      else if (k == 704) v = prevo0;
      else if (k < 715)  v = prevo[b * OUT_ + (k - 704)];   // slots 1..10, prev step
      else               v = 0.f;
      actF[b * KP_F + k] = (_Float16)v;
    }
#pragma unroll
    for (int off = 16; off > 0; off >>= 1) nsum += __shfl_xor(nsum, off, 32);
    if (lane == 0) padF[b * N_ + i] = nsum;
    __syncthreads();

    // ===== SD: final gates via WMMA =====
    gemm_gates<KT_F>(actF, KP_F, fh16, wfP, whfP, biasFL, gates, lane, w);
    __syncthreads();

    // ===== SE: final LSTM (regs) + output head + commit row =====
    float fh2[4];
#pragma unroll
    for (int q = 0; q < 4; ++q) {
      const int j = lane + 32 * q;
      const float ii = gates[b * G_ + j],       ff = gates[b * G_ + 128 + j];
      const float gg = gates[b * G_ + 256 + j], oo = gates[b * G_ + 384 + j];
      const float c2 = sigf(ff) * fc_r[q] + sigf(ii) * tanhf(gg);
      const float h2 = sigf(oo) * tanhf(c2);
      fc_r[q] = c2; fh2[q] = h2;
      fh16[b * 128 + j] = (_Float16)h2;
    }
    float po[D_];
#pragma unroll
    for (int o = 0; o < D_; ++o) {
      float p = 0.f;
#pragma unroll
      for (int q = 0; q < 4; ++q) p += fh2[q] * WfcL[o * 128 + lane + 32 * q];
      po[o] = p;
    }
#pragma unroll
    for (int off = 16; off > 0; off >>= 1) {
#pragma unroll
      for (int o = 0; o < D_; ++o) po[o] += __shfl_xor(po[o], off, 32);
    }
    if (lane == 0) {
      prevo[b * OUT_] = prevo0;
      out[(size_t)(b * N_ + i) * OUT_] = prevo0;
#pragma unroll
      for (int o = 0; o < D_; ++o) {
        const float vo = po[o] + bfcR[o];
        prevo[b * OUT_ + 1 + o] = vo;
        out[(size_t)(b * N_ + i) * OUT_ + 1 + o] = vo;
      }
    }
    __threadfence();
    __syncthreads();
  }
}

// zero rows flagged as padding (row sum == 0), using 4B flags instead of 2KB rows
__global__ void mask_pad(const float* __restrict__ padF, float* __restrict__ out) {
  const int idx = blockIdx.x * blockDim.x + threadIdx.x;
  if (idx >= B_ * N_) return;
  if (padF[idx] == 0.f) {
#pragma unroll
    for (int c = 0; c < OUT_; ++c) out[(size_t)idx * OUT_ + c] = 0.f;
  }
}

extern "C" void kernel_launch(void* const* d_in, const int* in_sizes, int n_in,
                              void* d_out, int out_size, void* d_ws, size_t ws_size,
                              hipStream_t stream) {
  const float* note_emb        = (const float*)d_in[0];
  const float* beat_emb        = (const float*)d_in[1];
  const float* measure_emb     = (const float*)d_in[2];
  const int*   beat_numbers    = (const int*)d_in[3];
  const int*   measure_numbers = (const int*)d_in[4];
  const float* Wa    = (const float*)d_in[5];
  const float* ba    = (const float*)d_in[6];
  const float* ctx   = (const float*)d_in[7];
  const float* Wih_t = (const float*)d_in[8];
  const float* Whh_t = (const float*)d_in[9];
  const float* bih_t = (const float*)d_in[10];
  const float* bhh_t = (const float*)d_in[11];
  const float* Wih_f = (const float*)d_in[12];
  const float* Whh_f = (const float*)d_in[13];
  const float* bih_f = (const float*)d_in[14];
  const float* bhh_f = (const float*)d_in[15];
  const float* W_fc  = (const float*)d_in[16];
  const float* b_fc  = (const float*)d_in[17];
  const float* W_tfc = (const float*)d_in[18];
  const float* b_tfc = (const float*)d_in[19];
  float* out = (float*)d_out;

  // workspace carve: packed f16 weights (~1.25 MB, L2-resident) + biases + attn vec + pad flags
  _Float16* wtP  = (_Float16*)d_ws;
  _Float16* whtP = wtP  + (size_t)KT_T * NT_ * 512;
  _Float16* wfP  = whtP + (size_t)KT_H * NT_ * 512;
  _Float16* whfP = wfP  + (size_t)KT_F * NT_ * 512;
  float* biasT = (float*)(whfP + (size_t)KT_H * NT_ * 512);
  float* biasF = biasT + 512;
  float* attnV = biasF + 512;
  float* padF  = attnV + 16;                   // 16*2048 flags

  pack_wmma_b<<<KT_T * NT_, 32, 0, stream>>>(Wih_t, wtP, 203);
  pack_wmma_b<<<KT_H * NT_, 32, 0, stream>>>(Whh_t, whtP, 128);
  pack_wmma_b<<<KT_F * NT_, 32, 0, stream>>>(Wih_f, wfP, 715);
  pack_wmma_b<<<KT_H * NT_, 32, 0, stream>>>(Whh_f, whfP, 128);
  prep_misc<<<5, 256, 0, stream>>>(bih_t, bhh_t, bih_f, bhh_f, Wa, ba, ctx,
                                   biasT, biasF, attnV);

  // dynamic LDS: gates/prevo/bcur/WfcL/WtfcL/biases (f32) + h16/fh16/actT/actF (f16)
  const size_t smemF = (size_t)(B_ * G_ + B_ * OUT_ + B_ * D_ + D_ * 128 + 128 + 2 * G_) * 4;
  const size_t smemH = (size_t)(2 * B_ * 128 + B_ * KP_T + B_ * KP_F) * 2;
  const size_t smem  = smemF + smemH;   // ~82.7 KB (< 320 KB/WGP on CDNA5)

  han_decoder_seq<<<1, 512, smem, stream>>>(
      note_emb, beat_emb, measure_emb, beat_numbers, measure_numbers,
      W_fc, b_fc, W_tfc, b_tfc, wtP, whtP, wfP, whfP, biasT, biasF, attnV, padF, out);

  mask_pad<<<(B_ * N_ + 255) / 256, 256, 0, stream>>>(padF, out);
}
